// GlimpseDecoder_7928509628563
// MI455X (gfx1250) — compile-verified, matrix operand order checked by reading, not run on hardware
//
#include <hip/hip_runtime.h>
#include <hip/hip_bf16.h>
#include <math.h>

#define B_SZ     512
#define N_NODES  200
#define D_EMB    128
#define H_HEADS  8
#define DK       16
#define PROJ_W   384
#define NEG_INF_F (-1e9f)
#define TANH_CLIP_F 10.0f

typedef __attribute__((ext_vector_type(16))) _Float16 v16h;
typedef __attribute__((ext_vector_type(8)))  float    v8f;

// ---------------------------------------------------------------------------
// Kernel 1: proj = emb @ W_node via WMMA f32<-f16 16x16x32.
// M = B*N = 102400 (6400 m-tiles), Ncols = 384 (6 groups of 64), K = 128.
// One wave computes a 16x64 slab: the A fragment for each K-chunk is reused
// across 4 WMMAs (16 WMMAs per wave total), amortizing operand loads.
// ---------------------------------------------------------------------------
__global__ __launch_bounds__(256) void proj_gemm_wmma(
    const float* __restrict__ emb,    // [B*N, 128]
    const float* __restrict__ Wnode,  // [128, 384]
    float* __restrict__ proj)         // [B*N, 384]
{
    const int wave  = threadIdx.x >> 5;
    const int lane  = threadIdx.x & 31;
    const int wtile = blockIdx.x * 8 + wave;   // 6400*6 = 38400 wave-tiles
    const int mtile = wtile / 6;
    const int ngrp  = wtile % 6;               // 64-column group
    const int row   = lane & 15;               // A row / B col / D col
    const int half  = lane >> 4;

    const float* Arow    = emb + (size_t)(mtile * 16 + row) * D_EMB;
    const int    ncol0   = ngrp * 64 + row;

    v8f acc[4] = {};
    #pragma unroll
    for (int kk = 0; kk < 4; ++kk) {
        const int k0 = kk * 32;
        // 16-bit A 16x32 layout: VGPR v<4 -> K = 8*half + 2v (+1);
        //                        VGPR v>=4 -> K = 16 + 8*half + 2(v-4) (+1).
        v16h af = {};
        #pragma unroll
        for (int v = 0; v < 8; ++v) {
            int kb = (v < 4) ? (8 * half + 2 * v) : (16 + 8 * half + 2 * (v - 4));
            af[2 * v]     = (_Float16)Arow[k0 + kb];
            af[2 * v + 1] = (_Float16)Arow[k0 + kb + 1];
        }
        #pragma unroll
        for (int nt = 0; nt < 4; ++nt) {
            v16h bf = {};
            #pragma unroll
            for (int v = 0; v < 8; ++v) {
                int kb = (v < 4) ? (8 * half + 2 * v) : (16 + 8 * half + 2 * (v - 4));
                const float* Bp = Wnode + (size_t)(k0 + kb) * PROJ_W + ncol0 + nt * 16;
                bf[2 * v]     = (_Float16)Bp[0];
                bf[2 * v + 1] = (_Float16)Bp[PROJ_W];
            }
            acc[nt] = __builtin_amdgcn_wmma_f32_16x16x32_f16(
                false, af, false, bf, (short)0, acc[nt], false, false);
        }
    }
    // D layout: lane col = lane&15, VGPR v -> row = v + 8*half
    #pragma unroll
    for (int nt = 0; nt < 4; ++nt) {
        #pragma unroll
        for (int v = 0; v < 8; ++v) {
            int m = mtile * 16 + v + 8 * half;
            proj[(size_t)m * PROJ_W + ncol0 + nt * 16] = acc[nt][v];
        }
    }
}

// ---------------------------------------------------------------------------
// Kernel 2: graph_emb = mean_n(emb), fixed_ctx = graph_emb @ W_fixed.
// One block (128 threads) per batch element.
// ---------------------------------------------------------------------------
__global__ __launch_bounds__(128) void ctx_precompute(
    const float* __restrict__ emb,
    const float* __restrict__ Wfixed,
    float* __restrict__ graph_emb,
    float* __restrict__ fixed_ctx)
{
    __shared__ float sGe[D_EMB];
    const int b = blockIdx.x, d = threadIdx.x;
    float s = 0.0f;
    for (int n = 0; n < N_NODES; ++n)
        s += emb[((size_t)b * N_NODES + n) * D_EMB + d];
    s *= (1.0f / (float)N_NODES);
    sGe[d] = s;
    graph_emb[(size_t)b * D_EMB + d] = s;
    __syncthreads();
    float f = 0.0f;
    for (int k = 0; k < D_EMB; ++k)
        f += sGe[k] * Wfixed[(size_t)k * D_EMB + d];
    fixed_ctx[(size_t)b * D_EMB + d] = f;
}

// ---------------------------------------------------------------------------
// Kernel 3: sequential 200-step decode. One block (8 wave32s) per batch
// element; gK/gV/lK for that element resident in 320KB LDS for all steps.
// ---------------------------------------------------------------------------
__global__ __launch_bounds__(256) void decode_kernel(
    const float* __restrict__ emb,       // [B,N,D]
    const float* __restrict__ Wstep,     // [D,D]
    const float* __restrict__ proj,      // [B*N, 384]
    const float* __restrict__ graph_emb, // [B,D]
    const float* __restrict__ fixed_ctx, // [B,D]
    float* __restrict__ out_logp,        // [B,T,N]
    float* __restrict__ out_pi)          // [B,T]
{
    extern __shared__ float smem[];
    float* sGK     = smem;               // 8*200*16 = 25600
    float* sGV     = sGK + 25600;        // 25600
    float* sLK     = sGV + 25600;        // 200*129 (padded) = 25800
    float* sFixed  = sLK + 25800;        // 128
    float* sCtx    = sFixed + 128;       // 128
    float* sQ      = sCtx + 128;         // 128
    float* sGl     = sQ + 128;           // 128
    float* sCompat = sGl + 128;          // 1600
    float* sLogits = sCompat + 1600;     // 200
    float* sRed    = sLogits + 200;      // 256
    int*   sVis    = (int*)(sRed + 256); // 200
    int*   sRedI   = sVis + 200;         // 8
    int*   sSel    = sRedI + 8;          // 1

    const int tid  = threadIdx.x;
    const int lane = tid & 31;
    const int w    = tid >> 5;
    const int b    = blockIdx.x;
    const float* projB = proj + (size_t)b * N_NODES * PROJ_W;

    // ---- stage per-batch K/V/logitK into LDS (once) ----
    for (int i = tid; i < H_HEADS * N_NODES * DK; i += 256) {
        int h = i / (N_NODES * DK);
        int r = i - h * N_NODES * DK;
        int n = r / DK;
        int k = r - n * DK;
        const float* p = projB + (size_t)n * PROJ_W + h * DK + k;
        sGK[i] = p[0];        // gK = proj[..., :128], [B,H,N,dk]
        sGV[i] = p[D_EMB];    // gV = proj[..., 128:256]
    }
    for (int i = tid; i < N_NODES * D_EMB; i += 256) {
        int n = i >> 7, d = i & 127;
        sLK[n * 129 + d] = projB[(size_t)n * PROJ_W + 256 + d]; // lK, padded
    }
    if (tid < D_EMB) {
        sFixed[tid] = fixed_ctx[(size_t)b * D_EMB + tid];
        sCtx[tid]   = graph_emb[(size_t)b * D_EMB + tid];
    }
    for (int n = tid; n < N_NODES; n += 256) sVis[n] = 0;

    // ---- W_step cached in registers: thread (d, half) keeps 64 weights ----
    const int dreg  = tid & 127;
    const int halfk = tid >> 7;
    float wreg[64];
    #pragma unroll
    for (int i = 0; i < 64; ++i)
        wreg[i] = Wstep[(size_t)(halfk * 64 + i) * D_EMB + dreg];

    __syncthreads();

    const float inv_sqrt_dk = 0.25f;                 // 1/sqrt(16)
    const float inv_sqrt_d  = 0.08838834764831845f;  // 1/sqrt(128)

    for (int t = 0; t < N_NODES; ++t) {
        // ---- q = fixed_ctx + ctx @ W_step ----
        float part = 0.0f;
        #pragma unroll
        for (int i = 0; i < 64; ++i) part += sCtx[halfk * 64 + i] * wreg[i];
        sRed[tid] = part;
        __syncthreads();
        if (tid < D_EMB) sQ[tid] = sFixed[tid] + sRed[tid] + sRed[tid + 128];
        __syncthreads();

        // ---- compat[h][n] = (qh . gK) / sqrt(dk), masked ----
        for (int i = tid; i < H_HEADS * N_NODES; i += 256) {
            int h = i / N_NODES;
            int n = i - h * N_NODES;
            const float* gk = sGK + i * DK;
            const float* qh = sQ + h * DK;
            float acc = 0.0f;
            #pragma unroll
            for (int k = 0; k < DK; ++k) acc += qh[k] * gk[k];
            sCompat[i] = sVis[n] ? NEG_INF_F : acc * inv_sqrt_dk;
        }
        __syncthreads();

        // ---- softmax per head: wave w owns head w ----
        {
            float m = -INFINITY;
            for (int n = lane; n < N_NODES; n += 32)
                m = fmaxf(m, sCompat[w * N_NODES + n]);
            #pragma unroll
            for (int off = 16; off > 0; off >>= 1)
                m = fmaxf(m, __shfl_xor(m, off, 32));
            float s = 0.0f;
            for (int n = lane; n < N_NODES; n += 32) {
                float e = __expf(sCompat[w * N_NODES + n] - m);
                sCompat[w * N_NODES + n] = e;
                s += e;
            }
            #pragma unroll
            for (int off = 16; off > 0; off >>= 1) s += __shfl_xor(s, off, 32);
            float inv = 1.0f / s;
            for (int n = lane; n < N_NODES; n += 32)
                sCompat[w * N_NODES + n] *= inv;
        }
        __syncthreads();

        // ---- glimpse[h*16+k] = sum_n attn * gV ----
        if (tid < D_EMB) {
            int h = tid >> 4, k = tid & 15;
            float acc = 0.0f;
            for (int n = 0; n < N_NODES; ++n)
                acc += sCompat[h * N_NODES + n] * sGV[(h * N_NODES + n) * DK + k];
            sGl[tid] = acc;
        }
        __syncthreads();

        // ---- logits[n] = clip*tanh((glimpse . lK)/sqrt(D)), masked ----
        if (tid < N_NODES) {
            const float* lk = sLK + tid * 129;
            float acc = 0.0f;
            #pragma unroll 8
            for (int d = 0; d < D_EMB; ++d) acc += sGl[d] * lk[d];
            float lg = TANH_CLIP_F * tanhf(acc * inv_sqrt_d);
            sLogits[tid] = sVis[tid] ? NEG_INF_F : lg;
        }
        __syncthreads();

        // ---- argmax + max (first-max tie-break) ----
        float v  = (tid < N_NODES) ? sLogits[tid] : -INFINITY;
        int   ix = (tid < N_NODES) ? tid : 0x7FFFFFFF;
        #pragma unroll
        for (int off = 16; off > 0; off >>= 1) {
            float ov = __shfl_xor(v, off, 32);
            int   oi = __shfl_xor(ix, off, 32);
            if (ov > v || (ov == v && oi < ix)) { v = ov; ix = oi; }
        }
        if (lane == 0) { sRed[w] = v; sRedI[w] = ix; }
        __syncthreads();
        if (tid == 0) {
            float bv = sRed[0]; int bi = sRedI[0];
            for (int i = 1; i < 8; ++i)
                if (sRed[i] > bv || (sRed[i] == bv && sRedI[i] < bi)) {
                    bv = sRed[i]; bi = sRedI[i];
                }
            sSel[0]  = bi;
            sRed[32] = bv;
            sVis[bi] = 1;
            out_pi[(size_t)b * N_NODES + t] = (float)bi;
        }
        __syncthreads();

        // ---- log_softmax over logits ----
        const float maxv = sRed[32];
        float e = (tid < N_NODES) ? __expf(sLogits[tid] - maxv) : 0.0f;
        #pragma unroll
        for (int off = 16; off > 0; off >>= 1) e += __shfl_xor(e, off, 32);
        if (lane == 0) sRed[64 + w] = e;
        __syncthreads();
        if (tid == 0) {
            float s = 0.0f;
            for (int i = 0; i < 8; ++i) s += sRed[64 + i];
            sRed[33] = __logf(s);
        }
        __syncthreads();
        const float lse = sRed[33];
        if (tid < N_NODES)
            out_logp[((size_t)b * N_NODES + t) * N_NODES + tid] =
                sLogits[tid] - maxv - lse;

        // ---- ctx_emb = embeddings[b, sel] ----
        const int sel = sSel[0];
        if (tid < D_EMB)
            sCtx[tid] = emb[((size_t)b * N_NODES + sel) * D_EMB + tid];
        __syncthreads();
    }
}

// ---------------------------------------------------------------------------
extern "C" void kernel_launch(void* const* d_in, const int* in_sizes, int n_in,
                              void* d_out, int out_size, void* d_ws, size_t ws_size,
                              hipStream_t stream) {
    (void)in_sizes; (void)n_in; (void)out_size; (void)ws_size;
    const float* emb    = (const float*)d_in[0];
    const float* Wnode  = (const float*)d_in[1];
    const float* Wfixed = (const float*)d_in[2];
    const float* Wstep  = (const float*)d_in[3];
    float* out = (float*)d_out;

    // Workspace layout: proj [B*N,384] | graph_emb [B,128] | fixed_ctx [B,128]
    float* proj      = (float*)d_ws;
    float* graph_emb = proj + (size_t)B_SZ * N_NODES * PROJ_W;
    float* fixed_ctx = graph_emb + (size_t)B_SZ * D_EMB;

    // 1) WMMA GEMM: 6400 m-tiles * 6 column groups, 8 waves per block
    proj_gemm_wmma<<<(6400 * 6) / 8, 256, 0, stream>>>(emb, Wnode, proj);

    // 2) graph context
    ctx_precompute<<<B_SZ, 128, 0, stream>>>(emb, Wfixed, graph_emb, fixed_ctx);

    // 3) sequential decode, one WGP-resident block per batch element
    const size_t smemBytes =
        (size_t)(25600 + 25600 + 25800 + 128 * 4 + 1600 + 200 + 256) * 4 +
        (size_t)(200 + 8 + 1) * 4;
    hipFuncSetAttribute((const void*)decode_kernel,
                        hipFuncAttributeMaxDynamicSharedMemorySize,
                        (int)smemBytes);
    float* out_logp = out;
    float* out_pi   = out + (size_t)B_SZ * N_NODES * N_NODES;
    decode_kernel<<<B_SZ, 256, smemBytes, stream>>>(
        emb, Wstep, proj, graph_emb, fixed_ctx, out_logp, out_pi);
}
